// SparseConv_37933151158306
// MI455X (gfx1250) — matrix operand order repaired and checked by reading, not compiled, not run on hardware
//
#include <hip/hip_runtime.h>

// ---------------------------------------------------------------------------
// CDNA5 (gfx1250) WMMA implementation of the sparse-conv GNN forward pass.
// All matmuls -> v_wmma_f32_16x16x32_f16 (f16 inputs, f32 accumulate).
// Fully templated GEMM (compile-time Kd/N/mode) -> unrolled straight-line
// WMMA chains, branch-free operand loads, LDS-staged weight panels.
// ---------------------------------------------------------------------------

typedef _Float16 v16h __attribute__((ext_vector_type(16)));
typedef _Float16 v8h  __attribute__((ext_vector_type(8)));
typedef float    v8f  __attribute__((ext_vector_type(8)));

#define BB 32
#define EE 2048
#define KK 16
#define BE (BB * EE)      // 65536 points
#define R1 (BE * KK)      // 1048576 gathered rows

enum { MODE_PLAIN_F32 = 0, MODE_PLAIN_F16 = 1, MODE_GATHER = 2, MODE_GATHER_WIDE = 3 };

// ---------------------------------------------------------------------------
// Generic templated GEMM: Y[M,N] = act(A[M,Kd] @ W[Kd,N] + bias).
//   MODE_PLAIN_F32 / F16 : A = X[row*Kd + k]
//   MODE_GATHER          : row->(be,k); A = feats[b, neigh[row], k] (* wf[row])
//   MODE_GATHER_WIDE     : row=be; k->(kn,f); A = feats[b, neigh[be,kn], f]
// 256 threads = 8 waves; each wave computes two 16x16 C tiles (256 rows/blk).
// W panel (Kd x 16) staged once in LDS as f16. No early returns: EXEC is
// all-ones at every WMMA. All loads are branch-free (clamped row + select).
// MFULL: M is an exact multiple of 256 -> all row guards/clamps elided.
// ---------------------------------------------------------------------------
template <int Kd, int N, int MODE, int F, bool SCALED, bool RELU, bool OUTF16,
          bool MFULL>
__global__ __launch_bounds__(256) void k_gemm(
    const void* __restrict__ X, const float* __restrict__ Wt,
    const float* __restrict__ bias, void* __restrict__ Y, int M,
    const int* __restrict__ neigh, const float* __restrict__ scale)
{
  static_assert(MODE != MODE_PLAIN_F16 || (Kd % 16 == 0), "f16 A needs Kd%16==0");
  constexpr int KP  = ((Kd + 31) / 32) * 32;   // padded K (chunks of 32)
  constexpr int NCH = KP / 32;                 // wmma chunks

  __shared__ __align__(64) _Float16 lds_w[16 * KP];   // [ncol][k], <=20KB

  const int tid  = threadIdx.x;
  const int wave = tid >> 5;
  const int lane = tid & 31;
  const int half = lane >> 4;      // 0: lanes 0-15, 1: lanes 16-31
  const int ln   = lane & 15;      // row (A) / col (B,C) within tile
  const int n0   = blockIdx.y * 16;
  const int m0   = blockIdx.x * 256 + wave * 32;   // 2 tiles of 16 rows

  // ---- Stage W panel [Kd x 16] -> f16 LDS (zero padded). Uniform trips. ----
  for (int i = tid; i < 16 * KP; i += 256) {
    int col = i / KP;
    int k   = i - col * KP;
    int gn  = n0 + col;
    float wv = (k < Kd && gn < N) ? Wt[k * N + gn] : 0.f;
    lds_w[col * KP + k] = (_Float16)wv;
  }
  __syncthreads();

  // ---- Per-tile row bookkeeping (branch-free row clamp; none if MFULL). ----
  int  rowc[2];
  long base[2];
  float sc[2];
#pragma unroll
  for (int t = 0; t < 2; ++t) {
    int row = m0 + t * 16 + ln;
    rowc[t] = MFULL ? row : (row < M ? row : M - 1);
    if constexpr (MODE == MODE_GATHER) {
      int b = rowc[t] >> 15;                  // row / (EE*KK)
      int j = neigh[rowc[t]];                 // neigh flat [BE*KK]
      base[t] = (long)(b * EE + j) * F;
      sc[t]   = SCALED ? scale[rowc[t]] : 1.f;
    } else if constexpr (MODE == MODE_GATHER_WIDE) {
      base[t] = 0;  sc[t] = 1.f;
    } else {
      base[t] = (long)rowc[t] * Kd;  sc[t] = 1.f;
    }
  }

  v8f acc[2] = {{0.f,0.f,0.f,0.f,0.f,0.f,0.f,0.f},
                {0.f,0.f,0.f,0.f,0.f,0.f,0.f,0.f}};

#pragma unroll
  for (int c = 0; c < NCH; ++c) {
    const int c32 = c * 32;
    // B fragment (same K order as A): e0-7 -> K c32+half*8+0-7,
    // e8-15 -> K c32+16+half*8+0-7. Two 16B vector ds loads.
    v8h b0 = *(const v8h*)&lds_w[ln * KP + c32 + half * 8];
    v8h b1 = *(const v8h*)&lds_w[ln * KP + c32 + 16 + half * 8];
    v16h vb;
#pragma unroll
    for (int e = 0; e < 8; ++e) { vb[e] = b0[e]; vb[e + 8] = b1[e]; }

#pragma unroll
    for (int t = 0; t < 2; ++t) {
      // A fragment, 16-bit A 16x32 layout:
      //  lanes 0-15 : e0-7 -> K c32+0-7,  e8-15 -> K c32+16-23
      //  lanes16-31 : e0-7 -> K c32+8-15, e8-15 -> K c32+24-31
      v16h a;
      if constexpr (MODE == MODE_PLAIN_F16) {
        const _Float16* Xh = (const _Float16*)X + base[t];
        // Plain 'if': loop-constant after unrolling, folds at -O3.
        if (c32 + 31 < Kd) {                  // full chunk: 2x 16B loads
          v8h g0 = *(const v8h*)(Xh + c32 + half * 8);
          v8h g1 = *(const v8h*)(Xh + c32 + 16 + half * 8);
#pragma unroll
          for (int e = 0; e < 8; ++e) { a[e] = g0[e]; a[e + 8] = g1[e]; }
        } else {                              // tail (Kd%32==16): low half only
          v8h g0 = *(const v8h*)(Xh + c32 + half * 8);
#pragma unroll
          for (int e = 0; e < 8; ++e) { a[e] = g0[e]; a[e + 8] = (_Float16)0.f; }
        }
      } else {
#pragma unroll
        for (int e = 0; e < 16; ++e) {
          int kl  = half * 8 + e + ((e >= 8) ? 8 : 0);
          int gk  = c32 + kl;
          int gkc = gk < Kd ? gk : Kd - 1;    // clamp -> unconditional load
          float v;
          if constexpr (MODE == MODE_PLAIN_F32) {
            v = ((const float*)X)[base[t] + gkc];
          } else if constexpr (MODE == MODE_GATHER) {
            v = ((const float*)X)[base[t] + gkc] * sc[t];
          } else {  // MODE_GATHER_WIDE
            int kn = gkc / F;
            int f  = gkc - kn * F;
            int b  = rowc[t] >> 11;           // be / EE
            int j  = neigh[rowc[t] * KK + kn];
            v = ((const float*)X)[(long)(b * EE + j) * F + f];
          }
          a[e] = (gk < Kd) ? (_Float16)v : (_Float16)0.f;   // select, no branch
        }
      }
      acc[t] = __builtin_amdgcn_wmma_f32_16x16x32_f16(
          /*neg_a=*/false, a, /*neg_b=*/false, vb,
          /*c_mod=*/(short)0, acc[t], /*reuse_a=*/false, /*reuse_b=*/false);
    }
  }

  // ---- Epilogue. C/D layout: VGPR r, lanes 0-15 -> M=r, 16-31 -> M=r+8. ----
  // Bias loaded ONCE per lane (clamped index); single col predicate wraps all
  // stores of a tile instead of per-element exec churn.
  const int  col = n0 + ln;
  const bool cok = col < N;
  const float bv = bias[cok ? col : 0];
#pragma unroll
  for (int t = 0; t < 2; ++t) {
    if (cok) {
#pragma unroll
      for (int r = 0; r < 8; ++r) {
        int rr = m0 + t * 16 + r + half * 8;
        bool rok = true;
        if constexpr (!MFULL) rok = rr < M;
        if (rok) {
          float y = acc[t][r] + bv;
          if constexpr (RELU) y = fmaxf(y, 0.f);
          if constexpr (OUTF16) ((_Float16*)Y)[(long)rr * N + col] = (_Float16)y;
          else                  ((float*)Y)[(long)rr * N + col]    = y;
        }
      }
    }
  }
}

template <int Kd, int N, int MODE, int F = 1, bool SCALED = false,
          bool RELU = false, bool OUTF16 = false, bool MFULL = false>
static inline void gemm(const void* X, const float* Wt, const float* b, void* Y,
                        int M, const int* neigh, const float* scale,
                        hipStream_t s)
{
  dim3 g((unsigned)((M + 255) / 256), (unsigned)((N + 15) / 16));
  k_gemm<Kd, N, MODE, F, SCALED, RELU, OUTF16, MFULL>
      <<<g, 256, 0, s>>>(X, Wt, b, Y, M, neigh, scale);
}

// ---------------------------------------------------------------------------
// Elementwise kernels.
// ---------------------------------------------------------------------------

// delta[be,k,s] = space[b, neigh[be,k], s] - space[be, s]   -> [BE, 48]
__global__ void k_delta(const float* __restrict__ space, const int* __restrict__ neigh,
                        float* __restrict__ delta)
{
  int i = blockIdx.x * blockDim.x + threadIdx.x;
  if (i >= BE * KK * 3) return;
  int s = i % 3, t = i / 3;
  int k = t & 15, be = t >> 4;
  int b = be / EE;
  int j = neigh[be * KK + k];
  delta[(long)be * 48 + k * 3 + s] =
      space[(long)(b * EE + j) * 3 + s] - space[(long)be * 3 + s];
}

// In-place row softmax over 16 + clip to [1e-5, 0.99999].
__global__ void k_softmax16(float* __restrict__ x, int rows)
{
  int r = blockIdx.x * blockDim.x + threadIdx.x;
  if (r >= rows) return;
  float v[16];
  float mx = -3.4e38f;
#pragma unroll
  for (int i = 0; i < 16; ++i) { v[i] = x[(long)r * 16 + i]; mx = fmaxf(mx, v[i]); }
  float s = 0.f;
#pragma unroll
  for (int i = 0; i < 16; ++i) { v[i] = __expf(v[i] - mx); s += v[i]; }
  float inv = 1.f / s;
#pragma unroll
  for (int i = 0; i < 16; ++i) {
    float w = fminf(fmaxf(v[i] * inv, 1e-5f), 0.99999f);
    x[(long)r * 16 + i] = w;
  }
}

// sp_out = space + mean_k(delta * ws)
__global__ void k_space_update(const float* __restrict__ space, const float* __restrict__ delta,
                               const float* __restrict__ wsw, float* __restrict__ out)
{
  int i = blockIdx.x * blockDim.x + threadIdx.x;
  if (i >= BE * 3) return;
  int s = i % 3, be = i / 3;
  float acc = 0.f;
#pragma unroll
  for (int k = 0; k < 16; ++k)
    acc += delta[(long)be * 48 + k * 3 + s] * wsw[(long)be * 16 + k];
  out[i] = space[i] + acc * (1.f / 16.f);
}

// Masked mean over entries, clipped to +-1e9. act: [BE, 40].
__global__ void k_pool(const float* __restrict__ act, const int* __restrict__ nent,
                       float* __restrict__ pooled)
{
  int i = blockIdx.x * blockDim.x + threadIdx.x;
  if (i >= BB * 40) return;
  int b = i / 40, n = i % 40;
  int ne = nent[b];                       // >= 1 by construction
  float s = 0.f;
  for (int e = 0; e < ne; ++e) s += act[(long)(b * EE + e) * 40 + n];
  pooled[i] = fminf(fmaxf(s / (float)ne, -1e9f), 1e9f);
}

// ---------------------------------------------------------------------------
// Host orchestration.
// ---------------------------------------------------------------------------
extern "C" void kernel_launch(void* const* d_in, const int* in_sizes, int n_in,
                              void* d_out, int out_size, void* d_ws, size_t ws_size,
                              hipStream_t stream)
{
  (void)in_sizes; (void)n_in; (void)out_size; (void)ws_size;

  const float* SPACE = (const float*)d_in[0];   // [B,E,3]
  const float* ALLF  = (const float*)d_in[1];   // [B,E,16]
  const int*   NEI   = (const int*)d_in[2];     // [B,E,K]
  const int*   NENT  = (const int*)d_in[3];     // [B,1]

  // Params: pytree leaves in alphabetical dict order.
  auto P = [&](int i) { return (const float*)d_in[i]; };
  const float *fc1W = P(4),  *fc1b = P(5),  *fc2W = P(6),  *fc2b = P(7);
  const float *fc3W = P(8),  *fc3b = P(9);
  struct Bare { const float *d1W, *d1b, *d2W, *d2b; };
  struct Sc2  { const float *d1W, *d1b, *d2W, *d2b, *wgW, *wgb, *wsW, *wsb; };
  Bare l1{P(10), P(11), P(12), P(13)};
  Bare l2{P(14), P(15), P(16), P(17)};
  Bare l3{P(18), P(19), P(20), P(21)};
  Sc2  l4{P(22), P(23), P(24), P(25), P(26), P(27), P(28), P(29)};
  Sc2  l5{P(30), P(31), P(32), P(33), P(34), P(35), P(36), P(37)};
  Sc2  l6{P(38), P(39), P(40), P(41), P(42), P(43), P(44), P(45)};

  // Workspace carve (256-byte aligned). Dominated by f16 'pre' (~84 MB).
  char* baseP = (char*)d_ws;
  size_t off = 0;
  auto carve = [&](size_t bytes) -> void* {
    off = (off + 255) & ~(size_t)255;
    void* p = baseP + off;
    off += bytes;
    return p;
  };
  float* actA   = (float*)carve((size_t)BE * 40 * 4);   // ping
  float* actB   = (float*)carve((size_t)BE * 40 * 4);   // pong
  float* spA    = (float*)carve((size_t)BE * 3 * 4);
  float* spB    = (float*)carve((size_t)BE * 3 * 4);
  void*  pre    =          carve((size_t)R1 * 40 * 2);  // f16 [R1, n]
  float* delta  = (float*)carve((size_t)BE * 48 * 4);
  float* wf     = (float*)carve((size_t)BE * 16 * 4);
  float* wsw    = (float*)carve((size_t)BE * 16 * 4);
  float* pooled = (float*)carve((size_t)32 * 40 * 4);
  float* h1     = (float*)carve((size_t)32 * 100 * 4);
  float* h2     = (float*)carve((size_t)32 * 100 * 4);

  const int EW = BE * KK * 3;   // delta elements

  // ---- l1..l3: bare sparse conv (gather fused into d1 A-operand) ----
  gemm<16, 15, MODE_GATHER, 16, false, true, true, true>(ALLF, l1.d1W, l1.d1b, pre, R1, NEI, nullptr, stream);
  gemm<240, 15, MODE_PLAIN_F16, 1, false, true, false, true>(pre, l1.d2W, l1.d2b, actA, BE, NEI, nullptr, stream);
  gemm<15, 20, MODE_GATHER, 15, false, true, true, true>(actA, l2.d1W, l2.d1b, pre, R1, NEI, nullptr, stream);
  gemm<320, 20, MODE_PLAIN_F16, 1, false, true, false, true>(pre, l2.d2W, l2.d2b, actB, BE, NEI, nullptr, stream);
  gemm<20, 25, MODE_GATHER, 20, false, true, true, true>(actB, l3.d1W, l3.d1b, pre, R1, NEI, nullptr, stream);
  gemm<400, 25, MODE_PLAIN_F16, 1, false, true, false, true>(pre, l3.d2W, l3.d2b, actA, BE, NEI, nullptr, stream);

  // ---- l4: sc2 (feats actA w=25, space SPACE -> spA, out actB w=30) ----
  k_delta<<<(EW + 255) / 256, 256, 0, stream>>>(SPACE, NEI, delta);
  gemm<48, 16, MODE_PLAIN_F32, 1, false, false, false, true>(delta, l4.wgW, l4.wgb, wf, BE, NEI, nullptr, stream);
  k_softmax16<<<(BE + 255) / 256, 256, 0, stream>>>(wf, BE);
  gemm<400, 16, MODE_GATHER_WIDE, 25, false, false, false, true>(actA, l4.wsW, l4.wsb, wsw, BE, NEI, nullptr, stream);
  k_softmax16<<<(BE + 255) / 256, 256, 0, stream>>>(wsw, BE);
  gemm<25, 30, MODE_GATHER, 25, true, true, true, true>(actA, l4.d1W, l4.d1b, pre, R1, NEI, wf, stream);
  gemm<480, 30, MODE_PLAIN_F16, 1, false, true, false, true>(pre, l4.d2W, l4.d2b, actB, BE, NEI, nullptr, stream);
  k_space_update<<<(BE * 3 + 255) / 256, 256, 0, stream>>>(SPACE, delta, wsw, spA);

  // ---- l5: sc2 (feats actB w=30, space spA -> spB, out actA w=35) ----
  k_delta<<<(EW + 255) / 256, 256, 0, stream>>>(spA, NEI, delta);
  gemm<48, 16, MODE_PLAIN_F32, 1, false, false, false, true>(delta, l5.wgW, l5.wgb, wf, BE, NEI, nullptr, stream);
  k_softmax16<<<(BE + 255) / 256, 256, 0, stream>>>(wf, BE);
  gemm<480, 16, MODE_GATHER_WIDE, 30, false, false, false, true>(actB, l5.wsW, l5.wsb, wsw, BE, NEI, nullptr, stream);
  k_softmax16<<<(BE + 255) / 256, 256, 0, stream>>>(wsw, BE);
  gemm<30, 35, MODE_GATHER, 30, true, true, true, true>(actB, l5.d1W, l5.d1b, pre, R1, NEI, wf, stream);
  gemm<560, 35, MODE_PLAIN_F16, 1, false, true, false, true>(pre, l5.d2W, l5.d2b, actA, BE, NEI, nullptr, stream);
  k_space_update<<<(BE * 3 + 255) / 256, 256, 0, stream>>>(spA, delta, wsw, spB);

  // ---- l6: sc2 (feats actA w=35, space spB; sp_out discarded -> skip ws) ----
  k_delta<<<(EW + 255) / 256, 256, 0, stream>>>(spB, NEI, delta);
  gemm<48, 16, MODE_PLAIN_F32, 1, false, false, false, true>(delta, l6.wgW, l6.wgb, wf, BE, NEI, nullptr, stream);
  k_softmax16<<<(BE + 255) / 256, 256, 0, stream>>>(wf, BE);
  gemm<35, 40, MODE_GATHER, 35, true, true, true, true>(actA, l6.d1W, l6.d1b, pre, R1, NEI, wf, stream);
  gemm<640, 40, MODE_PLAIN_F16, 1, false, true, false, true>(pre, l6.d2W, l6.d2b, actB, BE, NEI, nullptr, stream);

  // ---- masked mean pool + FC head (logits -> d_out, fp32 [32,4]) ----
  k_pool<<<(BB * 40 + 255) / 256, 256, 0, stream>>>(actB, NENT, pooled);
  gemm<40, 100, MODE_PLAIN_F32, 1, false, true, false, false>(pooled, fc1W, fc1b, h1, 32, NEI, nullptr, stream);
  gemm<100, 100, MODE_PLAIN_F32, 1, false, true, false, false>(h1, fc2W, fc2b, h2, 32, NEI, nullptr, stream);
  gemm<100, 4, MODE_PLAIN_F32, 1, false, false, false, false>(h2, fc3W, fc3b, d_out, 32, NEI, nullptr, stream);
}
